// MultiHeadedAttention_24919400251485
// MI455X (gfx1250) — matrix operand order repaired and checked by reading, not compiled, not run on hardware
//
#include <hip/hip_runtime.h>
#include <cstdint>
#include <cstddef>

// ---------------------------------------------------------------------------
// MultiHeadedAttention for MI455X (gfx1250), wave32 + WMMA bf16 path.
// B=4, T=2048, D=1024, H=16, DK=64.
// d_out = [ x : B*T*D floats ][ p_attn : B*H*T*T floats ]
// ---------------------------------------------------------------------------

#define BB 4
#define TT 2048
#define DD 1024
#define HH 16
#define DKK 64

typedef __attribute__((ext_vector_type(16))) __bf16 v16bf;
typedef __attribute__((ext_vector_type(8)))  __bf16 v8bf;
typedef __attribute__((ext_vector_type(8)))  float  v8f;

__device__ __forceinline__ __bf16 f2bf(float f) {
  union { float f; uint32_t u; } a; a.f = f;
  uint32_t r = a.u + 0x7FFFu + ((a.u >> 16) & 1u);   // round-to-nearest-even
  uint16_t h = (uint16_t)(r >> 16);
  return __builtin_bit_cast(__bf16, h);
}

__device__ __forceinline__ v16bf cat16(v8bf a, v8bf b) {
  return __builtin_shufflevector(a, b, 0,1,2,3,4,5,6,7,8,9,10,11,12,13,14,15);
}

__device__ __forceinline__ v8f bfwmma(v16bf a, v16bf b, v8f c) {
  // D = A(16x32 bf16) x B(32x16 bf16) + C(16x16 f32)
  return __builtin_amdgcn_wmma_f32_16x16x32_bf16(false, a, false, b,
                                                 (short)0, c, false, false);
}

// ---------------------------------------------------------------------------
// Generic tiled GEMM:  C(M=8192 x N=1024) = A(M x 1024) @ W(1024 x N) + bias
// Block tile 128(M) x 64(N), 8 waves of 32x32, K-chunks of 32 via LDS.
// IN_BF16: 0 -> A is fp32 (converted on stage-in), 1 -> A already bf16
// OUT_MODE: 0 -> bf16 row-major (M,1024)
//           1 -> bf16 transposed V layout (B,H,DK,T)
//           2 -> fp32 row-major (M,1024)
// ---------------------------------------------------------------------------
template <int IN_BF16, int OUT_MODE>
__global__ __launch_bounds__(256) void gemm_wmma(const void* __restrict__ Ain,
                                                 const float* __restrict__ Wg,
                                                 const float* __restrict__ bias,
                                                 void* __restrict__ Out) {
  __shared__ __attribute__((aligned(16))) __bf16 ldsA[128 * 40]; // 128 rows x 32k (+8 pad)
  __shared__ __attribute__((aligned(16))) __bf16 ldsW[64 * 40];  // 64 n-rows x 32k (+8 pad), transposed

  const int tid  = threadIdx.x;
  const int lane = tid & 31, wid = tid >> 5;
  const int wm = wid & 3, wn = wid >> 2;          // 4 x 2 wave grid
  const int half = lane >> 4, l15 = lane & 15;
  const size_t mblock = (size_t)blockIdx.x * 128;
  const int    nblock = blockIdx.y * 64;

  v8f acc[2][2] = {};
  float bsub[2];
  bsub[0] = bias[nblock + wn * 32 + l15];
  bsub[1] = bias[nblock + wn * 32 + 16 + l15];

  const int arow = tid >> 1;          // 0..127
  const int akp  = (tid & 1) * 16;    // 0 or 16
  const int wns  = tid & 63;          // n for W staging
  const int wk0  = (tid >> 6) * 8;    // 0,8,16,24

  for (int kb = 0; kb < 1024; kb += 32) {
    // ---- prefetch next K-chunk (lowered to global_prefetch_b8) ----
    if (kb + 32 < 1024) {
      if (IN_BF16)
        __builtin_prefetch((const __bf16*)Ain + (mblock + arow) * 1024 + kb + 32 + akp, 0, 1);
      else
        __builtin_prefetch((const float*)Ain + (mblock + arow) * 1024 + kb + 32 + akp, 0, 1);
      __builtin_prefetch(Wg + (size_t)(kb + 32 + wk0) * 1024 + nblock + wns, 0, 1);
    }
    // ---- stage A tile (row-major, bf16) ----
    if (IN_BF16) {
      const __bf16* src = (const __bf16*)Ain + (mblock + arow) * 1024 + kb + akp;
      v8bf x0 = *(const v8bf*)(src);
      v8bf x1 = *(const v8bf*)(src + 8);
      *(v8bf*)&ldsA[arow * 40 + akp]     = x0;
      *(v8bf*)&ldsA[arow * 40 + akp + 8] = x1;
    } else {
      const float* src = (const float*)Ain + (mblock + arow) * 1024 + kb + akp;
      v8bf y0, y1;
#pragma unroll
      for (int i = 0; i < 8; ++i) { y0[i] = f2bf(src[i]); y1[i] = f2bf(src[8 + i]); }
      *(v8bf*)&ldsA[arow * 40 + akp]     = y0;
      *(v8bf*)&ldsA[arow * 40 + akp + 8] = y1;
    }
    // ---- stage W tile transposed: ldsW[n][k] = W[kb+k][nblock+n] ----
    {
      const float* wsrc = Wg + (size_t)(kb + wk0) * 1024 + nblock + wns;
      v8bf wv;
#pragma unroll
      for (int i = 0; i < 8; ++i) wv[i] = f2bf(wsrc[(size_t)i * 1024]);
      *(v8bf*)&ldsW[wns * 40 + wk0] = wv;
    }
    __syncthreads();

#pragma unroll
    for (int sm = 0; sm < 2; ++sm) {
      const int row = wm * 32 + sm * 16 + l15;
      v8bf lo = *(const v8bf*)&ldsA[row * 40 + half * 8];
      v8bf hi = *(const v8bf*)&ldsA[row * 40 + 16 + half * 8];
      v16bf afrag = cat16(lo, hi);
#pragma unroll
      for (int sn = 0; sn < 2; ++sn) {
        const int nrow = wn * 32 + sn * 16 + l15;
        v8bf b0 = *(const v8bf*)&ldsW[nrow * 40 + half * 8];
        v8bf b1 = *(const v8bf*)&ldsW[nrow * 40 + 16 + half * 8];
        acc[sm][sn] = bfwmma(afrag, cat16(b0, b1), acc[sm][sn]);
      }
    }
    __syncthreads();
  }

  // ---- epilogue: C layout lane holds n=l15, rows m = 8*half + v ----
#pragma unroll
  for (int sm = 0; sm < 2; ++sm)
#pragma unroll
    for (int sn = 0; sn < 2; ++sn)
#pragma unroll
      for (int v = 0; v < 8; ++v) {
        const size_t m = mblock + wm * 32 + sm * 16 + 8 * half + v;
        const int    n = nblock + wn * 32 + sn * 16 + l15;
        const float val = acc[sm][sn][v] + bsub[sn];
        if (OUT_MODE == 0) {
          ((__bf16*)Out)[m * 1024 + n] = f2bf(val);
        } else if (OUT_MODE == 1) {
          const int b = (int)(m >> 11), t = (int)(m & 2047);
          const int h = n >> 6, dk = n & 63;
          ((__bf16*)Out)[(((size_t)b * HH + h) * DKK + dk) * TT + t] = f2bf(val);
        } else {
          ((float*)Out)[m * 1024 + n] = val;
        }
      }
}

// ---------------------------------------------------------------------------
// Scores: S[b,h,q,k] = (Q . K) / 8, written fp32 to P; online row max / sumexp
// stats written to Marr/Sarr.  Block = 8 waves x 16 q-rows = 128 q-rows.
// K tokens staged cooperatively in LDS (64 tokens x 64 dk per chunk) so all
// 8 waves share one global read of K.
// ---------------------------------------------------------------------------
__global__ __launch_bounds__(256) void attn_scores(const __bf16* __restrict__ Qbf,
                                                   const __bf16* __restrict__ Kbf,
                                                   float* __restrict__ P,
                                                   float* __restrict__ Marr,
                                                   float* __restrict__ Sarr) {
  __shared__ __attribute__((aligned(16))) __bf16 ldsK[64 * 72]; // 64 tokens x 64 dk (+8 pad)
  const int tid = threadIdx.x, lane = tid & 31, wid = tid >> 5;
  const int half = lane >> 4, l15 = lane & 15;
  const int bh = blockIdx.y, b = bh >> 4, h = bh & 15;
  const int qbase = blockIdx.x * 128 + wid * 16;

  // Q A-fragments for 16 rows, DK=64 -> two 16x32 fragments, held in regs.
  const __bf16* qptr = Qbf + ((size_t)(b * TT + qbase + l15)) * DD + h * DKK;
  v16bf a0 = cat16(*(const v8bf*)(qptr + half * 8),
                   *(const v8bf*)(qptr + 16 + half * 8));
  v16bf a1 = cat16(*(const v8bf*)(qptr + 32 + half * 8),
                   *(const v8bf*)(qptr + 48 + half * 8));

  float lm[8], ls[8];
#pragma unroll
  for (int v = 0; v < 8; ++v) { lm[v] = -1.0e30f; ls[v] = 0.0f; }

  float* prow = P + ((size_t)bh * TT + qbase) * TT;

  const int kr  = tid >> 2;          // staging token row 0..63
  const int ksg = (tid & 3) * 16;    // staging dk segment 0,16,32,48

  for (int nb = 0; nb < TT; nb += 64) {
    // ---- cooperative stage of K tile: 64 tokens x 64 dk ----
    {
      const __bf16* ksrc = Kbf + ((size_t)(b * TT + nb + kr)) * DD + h * DKK + ksg;
      v8bf k0 = *(const v8bf*)(ksrc);
      v8bf k1 = *(const v8bf*)(ksrc + 8);
      *(v8bf*)&ldsK[kr * 72 + ksg]     = k0;
      *(v8bf*)&ldsK[kr * 72 + ksg + 8] = k1;
      if (nb + 64 < TT)
        __builtin_prefetch(ksrc + (size_t)64 * DD, 0, 1);  // next token tile
    }
    __syncthreads();

#pragma unroll
    for (int sn = 0; sn < 4; ++sn) {
      const int trow = sn * 16 + l15;            // token row in LDS tile
      v16bf b0 = cat16(*(const v8bf*)&ldsK[trow * 72 + half * 8],
                       *(const v8bf*)&ldsK[trow * 72 + 16 + half * 8]);
      v16bf b1 = cat16(*(const v8bf*)&ldsK[trow * 72 + 32 + half * 8],
                       *(const v8bf*)&ldsK[trow * 72 + 48 + half * 8]);
      v8f c = {};
      c = bfwmma(a0, b0, c);
      c = bfwmma(a1, b1, c);
#pragma unroll
      for (int v = 0; v < 8; ++v) {
        const float s = c[v] * 0.125f;                       // 1/sqrt(64)
        prow[(size_t)(8 * half + v) * TT + nb + sn * 16 + l15] = s;
        const float nm = fmaxf(lm[v], s);
        ls[v] = ls[v] * __expf(lm[v] - nm) + __expf(s - nm);
        lm[v] = nm;
      }
    }
    __syncthreads();
  }

  // combine (max,sumexp) across the 16 lanes sharing a row set
#pragma unroll
  for (int v = 0; v < 8; ++v) {
#pragma unroll
    for (int mask = 1; mask < 16; mask <<= 1) {
      const float om = __shfl_xor(lm[v], mask, 32);
      const float os = __shfl_xor(ls[v], mask, 32);
      const float nm = fmaxf(lm[v], om);
      ls[v] = ls[v] * __expf(lm[v] - nm) + os * __expf(om - nm);
      lm[v] = nm;
    }
  }
  if (l15 == 0) {
#pragma unroll
    for (int v = 0; v < 8; ++v) {
      const int q = qbase + 8 * half + v;
      Marr[(size_t)bh * TT + q] = lm[v];
      Sarr[(size_t)bh * TT + q] = ls[v];
    }
  }
}

// ---------------------------------------------------------------------------
// Normalize p = exp(s-M)/S in place (fp32), and compute x' = P @ V via WMMA,
// re-shaping p C-layout -> A-fragment layout through a wave-private LDS tile.
// Block = 4 waves x 16 q-rows.
// ---------------------------------------------------------------------------
__global__ __launch_bounds__(128) void attn_pv(float* __restrict__ P,
                                               const float* __restrict__ Marr,
                                               const float* __restrict__ Sarr,
                                               const __bf16* __restrict__ VT,
                                               __bf16* __restrict__ AOut) {
  __shared__ __attribute__((aligned(16))) __bf16 ldsP[4][16 * 40];
  const int tid = threadIdx.x, lane = tid & 31, wid = tid >> 5;
  const int half = lane >> 4, l15 = lane & 15;
  const int bh = blockIdx.y, b = bh >> 4, h = bh & 15;
  const int qbase = blockIdx.x * 64 + wid * 16;
  __bf16* tile = &ldsP[wid][0];

  float rowM[8], rowSi[8];
#pragma unroll
  for (int v = 0; v < 8; ++v) {
    const int q = qbase + 8 * half + v;
    rowM[v]  = Marr[(size_t)bh * TT + q];
    rowSi[v] = 1.0f / Sarr[(size_t)bh * TT + q];
  }

  v8f acc[4] = {};
  float* prow = P + ((size_t)bh * TT + qbase) * TT;
  const __bf16* vbase = VT + (size_t)bh * DKK * TT;

  for (int kc = 0; kc < TT; kc += 32) {
#pragma unroll
    for (int j = 0; j < 2; ++j)
#pragma unroll
      for (int v = 0; v < 8; ++v) {
        const size_t idx = (size_t)(8 * half + v) * TT + kc + j * 16 + l15;
        const float s = prow[idx];
        const float p = __expf(s - rowM[v]) * rowSi[v];
        prow[idx] = p;                                     // final p_attn
        tile[(8 * half + v) * 40 + j * 16 + l15] = f2bf(p);
      }
    // wave-local LDS visibility (lanes cross-read); DS ops are in-order so
    // next iteration's stores cannot pass this iteration's reads.
    asm volatile("s_wait_dscnt 0" ::: "memory");

    v16bf afrag = cat16(*(const v8bf*)&tile[l15 * 40 + half * 8],
                        *(const v8bf*)&tile[l15 * 40 + 16 + half * 8]);
#pragma unroll
    for (int t = 0; t < 4; ++t) {
      const __bf16* vp = vbase + (size_t)(t * 16 + l15) * TT + kc + half * 8;
      v16bf bfrag = cat16(*(const v8bf*)vp, *(const v8bf*)(vp + 16));
      acc[t] = bfwmma(afrag, bfrag, acc[t]);
    }
  }

#pragma unroll
  for (int t = 0; t < 4; ++t)
#pragma unroll
    for (int v = 0; v < 8; ++v) {
      const int q = qbase + 8 * half + v;
      AOut[((size_t)(b * TT + q)) * DD + h * DKK + t * 16 + l15] = f2bf(acc[t][v]);
    }
}

// ---------------------------------------------------------------------------
extern "C" void kernel_launch(void* const* d_in, const int* in_sizes, int n_in,
                              void* d_out, int out_size, void* d_ws, size_t ws_size,
                              hipStream_t stream) {
  const float* query = (const float*)d_in[0];
  const float* key_  = (const float*)d_in[1];
  const float* value = (const float*)d_in[2];
  const float* Wq = (const float*)d_in[3];
  const float* bq = (const float*)d_in[4];
  const float* Wk = (const float*)d_in[5];
  const float* bk = (const float*)d_in[6];
  const float* Wv = (const float*)d_in[7];
  const float* bv = (const float*)d_in[8];
  const float* Wo = (const float*)d_in[9];
  const float* bo = (const float*)d_in[10];

  float* xout = (float*)d_out;                       // (B,T,D)
  float* pout = xout + (size_t)BB * TT * DD;         // (B,H,T,T)

  char* ws = (char*)d_ws;
  __bf16* Qbf  = (__bf16*)(ws);                      // 16 MiB (B,T,D) bf16
  __bf16* Kbf  = (__bf16*)(ws + (16ull << 20));      // 16 MiB (B,T,D) bf16
  __bf16* VT   = (__bf16*)(ws + (32ull << 20));      // 16 MiB (B,H,DK,T) bf16
  __bf16* AOut = (__bf16*)(ws + (48ull << 20));      // 16 MiB (B,T,D) bf16
  float*  Marr = (float*)(ws + (64ull << 20));       // 512 KiB (B,H,T)
  float*  Sarr = Marr + (size_t)BB * HH * TT;        // 512 KiB (B,H,T)

  const dim3 gg(64, 16), bg(256);
  // QKV projections (bf16 WMMA, fp32 accumulate)
  gemm_wmma<0, 0><<<gg, bg, 0, stream>>>(query, Wq, bq, Qbf);
  gemm_wmma<0, 0><<<gg, bg, 0, stream>>>(key_,  Wk, bk, Kbf);
  gemm_wmma<0, 1><<<gg, bg, 0, stream>>>(value, Wv, bv, VT);   // transposed V
  // scores + online softmax stats
  attn_scores<<<dim3(TT / 128, BB * HH), 256, 0, stream>>>(Qbf, Kbf, pout, Marr, Sarr);
  // normalize p in place + P@V
  attn_pv<<<dim3(TT / 64, BB * HH), 128, 0, stream>>>(pout, Marr, Sarr, VT, AOut);
  // output projection -> fp32 x
  gemm_wmma<1, 2><<<gg, bg, 0, stream>>>(AOut, Wo, bo, xout);
}
